// RWKV_Tmix_x060_86174223827666
// MI455X (gfx1250) — compile-verified, matrix operand order checked by reading, not run on hardware
//
#include <hip/hip_runtime.h>
#include <hip/hip_bf16.h>

// ---------------- problem constants ----------------
#define Bc   2
#define Tc   2048
#define Cc   2048
#define Hc   32
#define Nc   64
#define BTc  (Bc * Tc)          // 4096
#define BTCc ((size_t)BTc * Cc) // 8,388,608
#define DMIX 32
#define DLORA (5 * DMIX)        // 160
#define DDEC 64
#define EPSc 6.4e-4f            // 1e-5 * 8^2

typedef __attribute__((ext_vector_type(16))) __bf16 v16bf;
typedef __attribute__((ext_vector_type(8)))  float  v8f;

// ================= token shift =================
__global__ __launch_bounds__(256) void k_prep_xx(const float* __restrict__ x,
                                                 float* __restrict__ xx) {
  size_t idx = (size_t)blockIdx.x * 256 + threadIdx.x;
  if (idx >= BTCc) return;
  int row = (int)(idx / Cc);
  int t = row % Tc;
  float xc = x[idx];
  float xp = (t > 0) ? x[idx - Cc] : 0.f;
  xx[idx] = xp - xc;
}

// ============ LoRA stage 1: z = tanh(xxx @ w1), [BT,160] ============
__global__ __launch_bounds__(DLORA) void k_lora_z(const float* __restrict__ x,
                                                  const float* __restrict__ xx,
                                                  const float* __restrict__ maa_x,
                                                  const float* __restrict__ w1,
                                                  float* __restrict__ z) {
  int row = blockIdx.x;      // 0..BT-1
  int j = threadIdx.x;       // 0..159
  const float* xr = x + (size_t)row * Cc;
  const float* xxr = xx + (size_t)row * Cc;
  float acc = 0.f;
  for (int c = 0; c < Cc; ++c) {
    float xxx = xr[c] + xxr[c] * maa_x[c];
    acc += xxx * w1[(size_t)c * DLORA + j];
  }
  z[(size_t)row * DLORA + j] = tanhf(acc);
}

// ============ LoRA stage 2 + mixing: build xw/xk/xv/xr/xg (bf16) ============
__global__ __launch_bounds__(256) void k_mix(const float* __restrict__ x,
                                             const float* __restrict__ xx,
                                             const float* __restrict__ z,
                                             const float* __restrict__ maa_w,
                                             const float* __restrict__ maa_k,
                                             const float* __restrict__ maa_v,
                                             const float* __restrict__ maa_r,
                                             const float* __restrict__ maa_g,
                                             const float* __restrict__ w2,
                                             __bf16* __restrict__ oxw,
                                             __bf16* __restrict__ oxk,
                                             __bf16* __restrict__ oxv,
                                             __bf16* __restrict__ oxr,
                                             __bf16* __restrict__ oxg) {
  int row = blockIdx.x;
  int c = blockIdx.y * 256 + threadIdx.x;
  __shared__ float sz[DLORA];
  if (threadIdx.x < DLORA) sz[threadIdx.x] = z[(size_t)row * DLORA + threadIdx.x];
  __syncthreads();
  float m[5];
#pragma unroll
  for (int f = 0; f < 5; ++f) {
    float acc = 0.f;
#pragma unroll
    for (int d = 0; d < DMIX; ++d)
      acc += sz[f * DMIX + d] * w2[(size_t)(f * DMIX + d) * Cc + c];
    m[f] = acc;
  }
  size_t idx = (size_t)row * Cc + c;
  float xc = x[idx], xxc = xx[idx];
  oxw[idx] = (__bf16)(xc + xxc * (maa_w[c] + m[0]));
  oxk[idx] = (__bf16)(xc + xxc * (maa_k[c] + m[1]));
  oxv[idx] = (__bf16)(xc + xxc * (maa_v[c] + m[2]));
  oxr[idx] = (__bf16)(xc + xxc * (maa_r[c] + m[3]));
  oxg[idx] = (__bf16)(xc + xxc * (maa_g[c] + m[4]));
}

// ============ weight conversions ============
__global__ __launch_bounds__(256) void k_cvt5(const float* s0, const float* s1,
                                              const float* s2, const float* s3,
                                              const float* s4, __bf16* d0, __bf16* d1,
                                              __bf16* d2, __bf16* d3, __bf16* d4,
                                              size_t n) {
  const float* s;
  __bf16* d;
  switch (blockIdx.y) {
    case 0: s = s0; d = d0; break;
    case 1: s = s1; d = d1; break;
    case 2: s = s2; d = d2; break;
    case 3: s = s3; d = d3; break;
    default: s = s4; d = d4; break;
  }
  size_t i = (size_t)blockIdx.x * 256 + threadIdx.x;
  if (i < n) d[i] = (__bf16)s[i];
}

// time_decay_w1 [C,64] -> transposed bf16 [64,C]
__global__ __launch_bounds__(256) void k_t_w1(const float* __restrict__ w1,
                                              __bf16* __restrict__ out) {
  int idx = blockIdx.x * 256 + threadIdx.x;  // over 64*C
  int j = idx / Cc, c = idx % Cc;
  out[idx] = (__bf16)w1[(size_t)c * DDEC + j];
}

// ============ WMMA GEMM: Y[M,N] (f32) = A[M,K] (bf16) @ W[N,K]^T (bf16) ============
// block = 8 waves (2 M x 4 N); each wave owns a 64x64 tile = 4x4
// v_wmma_f32_16x16x32_bf16 accumulators -> 8 fragment loads feed 16 WMMAs per
// K-step (0.5 KB operand traffic per 16-KFLOP WMMA). Block tile = 128 x 256.
union FragU { uint4 u[2]; v16bf v; };

__global__ __launch_bounds__(256) void k_gemm_bf16(const __bf16* __restrict__ A,
                                                   const __bf16* __restrict__ W,
                                                   float* __restrict__ Y, int M,
                                                   int N, int K) {
  int lane = threadIdx.x & 31;
  int wave = threadIdx.x >> 5;
  int m0 = blockIdx.x * 128 + (wave >> 2) * 64;
  int n0 = blockIdx.y * 256 + (wave & 3) * 64;
  int lr = lane & 15;
  int khalf = (lane >> 4) * 8;  // lanes 0-15: K {0..7,16..23}; 16-31: {8..15,24..31}

  // per-lane base pointers (row fixed per fragment slot)
  const __bf16* pA[4];
  const __bf16* pB[4];
#pragma unroll
  for (int im = 0; im < 4; ++im)
    pA[im] = A + (size_t)(m0 + im * 16 + lr) * K + khalf;
#pragma unroll
  for (int in = 0; in < 4; ++in) {
    int rowc = n0 + in * 16 + lr;
    if (rowc >= N) rowc = N - 1;  // clamp for partial N tiles
    pB[in] = W + (size_t)rowc * K + khalf;
  }

  v8f acc[4][4] = {};
  for (int kk = 0; kk < K; kk += 32) {
    FragU a[4], b[4];
#pragma unroll
    for (int im = 0; im < 4; ++im) {
      a[im].u[0] = *(const uint4*)(pA[im] + kk);
      a[im].u[1] = *(const uint4*)(pA[im] + kk + 16);
    }
#pragma unroll
    for (int in = 0; in < 4; ++in) {
      b[in].u[0] = *(const uint4*)(pB[in] + kk);
      b[in].u[1] = *(const uint4*)(pB[in] + kk + 16);
    }
    // keep the L2->WGP stream ahead of the clause loads (global_prefetch_b8)
    __builtin_prefetch(pA[0] + kk + 128, 0, 0);
    __builtin_prefetch(pB[0] + kk + 128, 0, 0);
#pragma unroll
    for (int im = 0; im < 4; ++im)
#pragma unroll
      for (int in = 0; in < 4; ++in)
        acc[im][in] = __builtin_amdgcn_wmma_f32_16x16x32_bf16(
            false, a[im].v, false, b[in].v, (short)0, acc[im][in], false, false);
  }
  int nlane = lane & 15;
  int moff = (lane >> 4) * 8;  // C/D layout: lanes 16-31 hold M rows +8
#pragma unroll
  for (int im = 0; im < 4; ++im)
#pragma unroll
    for (int in = 0; in < 4; ++in) {
      int nn = n0 + in * 16 + nlane;
      if (nn >= N) continue;
#pragma unroll
      for (int r = 0; r < 8; ++r)
        Y[(size_t)(m0 + im * 16 + moff + r) * N + nn] = acc[im][in][r];
    }
}

// ============ small elementwise epilogues ============
__global__ __launch_bounds__(256) void k_silu_ip(float* __restrict__ g, size_t n) {
  size_t i = (size_t)blockIdx.x * 256 + threadIdx.x;
  if (i < n) {
    float v = g[i];
    g[i] = v / (1.f + __expf(-v));
  }
}
__global__ __launch_bounds__(256) void k_tanh_ip(float* __restrict__ d, size_t n) {
  size_t i = (size_t)blockIdx.x * 256 + threadIdx.x;
  if (i < n) d[i] = tanhf(d[i]);
}

// ww = time_decay + dec[BT,64] @ w2d[64,C]
__global__ __launch_bounds__(256) void k_ww(const float* __restrict__ dec,
                                            const float* __restrict__ w2d,
                                            const float* __restrict__ tdecay,
                                            float* __restrict__ ww) {
  int row = blockIdx.x;
  int c = blockIdx.y * 256 + threadIdx.x;
  __shared__ float sd[DDEC];
  if (threadIdx.x < DDEC) sd[threadIdx.x] = dec[(size_t)row * DDEC + threadIdx.x];
  __syncthreads();
  float acc = tdecay[c];
#pragma unroll 16
  for (int j = 0; j < DDEC; ++j) acc += sd[j] * w2d[(size_t)j * Cc + c];
  ww[(size_t)row * Cc + c] = acc;
}

// ============ WKV6 sequential recurrence ============
// one block per (b,h); thread j owns value column j; state[i][j] in registers.
__global__ __launch_bounds__(64) void k_wkv6(const float* __restrict__ r,
                                             const float* __restrict__ k,
                                             const float* __restrict__ v,
                                             const float* __restrict__ w,
                                             const float* __restrict__ u,
                                             float* __restrict__ out) {
  int bh = blockIdx.x;
  int b = bh / Hc, h = bh % Hc;
  int j = threadIdx.x;
  __shared__ float sr[Nc], sk[Nc], sw[Nc], su[Nc];
  su[j] = u[h * Nc + j];
  float state[Nc];
#pragma unroll
  for (int i = 0; i < Nc; ++i) state[i] = 0.f;

  for (int t = 0; t < Tc; ++t) {
    size_t base = ((size_t)(b * Tc + t)) * Cc + (size_t)h * Nc;
    __syncthreads();  // protect previous iteration's reads
    sr[j] = r[base + j];
    sk[j] = k[base + j];
    sw[j] = __expf(-__expf(w[base + j]));
    __syncthreads();
    float vj = v[base + j];
    float o = 0.f;
#pragma unroll
    for (int i = 0; i < Nc; ++i) {
      float kv = sk[i] * vj;
      o += sr[i] * (state[i] + su[i] * kv);
      state[i] = state[i] * sw[i] + kv;
    }
    out[base + j] = o;
  }
}

// ============ per-head GroupNorm + SiLU gate, emit bf16 for final GEMM ============
__global__ __launch_bounds__(256) void k_gnorm_gate(const float* __restrict__ xo,
                                                    const float* __restrict__ g,
                                                    const float* __restrict__ lnw,
                                                    const float* __restrict__ lnb,
                                                    __bf16* __restrict__ og) {
  int wave = threadIdx.x >> 5, lane = threadIdx.x & 31;
  int gid = blockIdx.x * 8 + wave;  // head-instance id in [0, BT*H)
  int row = gid / Hc, h = gid % Hc;
  size_t base = (size_t)row * Cc + (size_t)h * Nc;
  float a0 = xo[base + lane], a1 = xo[base + 32 + lane];
  float s = a0 + a1, ss = a0 * a0 + a1 * a1;
#pragma unroll
  for (int m = 16; m > 0; m >>= 1) {
    s += __shfl_xor(s, m, 32);
    ss += __shfl_xor(ss, m, 32);
  }
  float mean = s * (1.f / Nc);
  float inv = rsqrtf(ss * (1.f / Nc) - mean * mean + EPSc);
  int c0 = h * Nc + lane, c1 = c0 + 32;
  float y0 = ((a0 - mean) * inv * lnw[c0] + lnb[c0]) * g[base + lane];
  float y1 = ((a1 - mean) * inv * lnw[c1] + lnb[c1]) * g[base + 32 + lane];
  og[base + lane] = (__bf16)y0;
  og[base + 32 + lane] = (__bf16)y1;
}

// =================== host-side launch ===================
extern "C" void kernel_launch(void* const* d_in, const int* in_sizes, int n_in,
                              void* d_out, int out_size, void* d_ws, size_t ws_size,
                              hipStream_t stream) {
  const float* x       = (const float*)d_in[0];
  const float* maa_x   = (const float*)d_in[1];
  const float* maa_w   = (const float*)d_in[2];
  const float* maa_k   = (const float*)d_in[3];
  const float* maa_v   = (const float*)d_in[4];
  const float* maa_r   = (const float*)d_in[5];
  const float* maa_g   = (const float*)d_in[6];
  const float* maa_w1  = (const float*)d_in[7];
  const float* maa_w2  = (const float*)d_in[8];
  const float* tdecay  = (const float*)d_in[9];
  const float* dec_w1  = (const float*)d_in[10];
  const float* dec_w2  = (const float*)d_in[11];
  const float* faaaa   = (const float*)d_in[12];
  const float* W_r     = (const float*)d_in[13];
  const float* W_k     = (const float*)d_in[14];
  const float* W_v     = (const float*)d_in[15];
  const float* W_o     = (const float*)d_in[16];
  const float* W_g     = (const float*)d_in[17];
  const float* ln_w    = (const float*)d_in[18];
  const float* ln_b    = (const float*)d_in[19];
  float* out = (float*)d_out;

  char* ws = (char*)d_ws;
  size_t off = 0;
  auto alloc = [&](size_t bytes) -> void* {
    off = (off + 255) & ~(size_t)255;
    void* p = ws + off;
    off += bytes;
    return p;
  };
  float*  xx_f   = (float*)alloc(BTCc * 4);
  float*  z_f    = (float*)alloc((size_t)BTc * DLORA * 4);
  float*  dec_f  = (float*)alloc((size_t)BTc * DDEC * 4);
  __bf16* xw_bf  = (__bf16*)alloc(BTCc * 2);
  __bf16* xk_bf  = (__bf16*)alloc(BTCc * 2);
  __bf16* xv_bf  = (__bf16*)alloc(BTCc * 2);
  __bf16* xr_bf  = (__bf16*)alloc(BTCc * 2);
  __bf16* xg_bf  = (__bf16*)alloc(BTCc * 2);
  __bf16* Wr_bf  = (__bf16*)alloc((size_t)Cc * Cc * 2);
  __bf16* Wk_bf  = (__bf16*)alloc((size_t)Cc * Cc * 2);
  __bf16* Wv_bf  = (__bf16*)alloc((size_t)Cc * Cc * 2);
  __bf16* Wg_bf  = (__bf16*)alloc((size_t)Cc * Cc * 2);
  __bf16* Wo_bf  = (__bf16*)alloc((size_t)Cc * Cc * 2);
  __bf16* w1T_bf = (__bf16*)alloc((size_t)DDEC * Cc * 2);
  float*  r_f    = (float*)alloc(BTCc * 4);
  float*  k_f    = (float*)alloc(BTCc * 4);
  float*  v_f    = (float*)alloc(BTCc * 4);
  float*  ww_f   = (float*)alloc(BTCc * 4);
  float*  g_f    = (float*)alloc(BTCc * 4);
  float*  wkv_f  = (float*)alloc(BTCc * 4);
  __bf16* og_bf  = (__bf16*)alloc(BTCc * 2);
  (void)ws_size; (void)in_sizes; (void)n_in; (void)out_size;

  // 1) token shift
  k_prep_xx<<<dim3((unsigned)(BTCc / 256)), 256, 0, stream>>>(x, xx_f);
  // 2) LoRA stage 1
  k_lora_z<<<BTc, DLORA, 0, stream>>>(x, xx_f, maa_x, maa_w1, z_f);
  // 3) weight conversions (concurrent with 1/2)
  k_cvt5<<<dim3((unsigned)(((size_t)Cc * Cc) / 256), 5), 256, 0, stream>>>(
      W_r, W_k, W_v, W_g, W_o, Wr_bf, Wk_bf, Wv_bf, Wg_bf, Wo_bf, (size_t)Cc * Cc);
  k_t_w1<<<(DDEC * Cc) / 256, 256, 0, stream>>>(dec_w1, w1T_bf);
  // 4) LoRA stage 2 + build mixed bf16 activations
  k_mix<<<dim3(BTc, Cc / 256), 256, 0, stream>>>(x, xx_f, z_f, maa_w, maa_k, maa_v,
                                                 maa_r, maa_g, maa_w2, xw_bf, xk_bf,
                                                 xv_bf, xr_bf, xg_bf);
  // 5) big WMMA GEMMs: r/k/v/g projections (block tile 128x256)
  dim3 gg(BTc / 128, Cc / 256);
  k_gemm_bf16<<<gg, 256, 0, stream>>>(xr_bf, Wr_bf, r_f, BTc, Cc, Cc);
  k_gemm_bf16<<<gg, 256, 0, stream>>>(xk_bf, Wk_bf, k_f, BTc, Cc, Cc);
  k_gemm_bf16<<<gg, 256, 0, stream>>>(xv_bf, Wv_bf, v_f, BTc, Cc, Cc);
  k_gemm_bf16<<<gg, 256, 0, stream>>>(xg_bf, Wg_bf, g_f, BTc, Cc, Cc);
  // 6) decay path: dec = tanh(xw @ w1) via WMMA (N=64, partial tile), then expand
  k_gemm_bf16<<<dim3(BTc / 128, 1), 256, 0, stream>>>(xw_bf, w1T_bf, dec_f, BTc, DDEC, Cc);
  k_tanh_ip<<<(BTc * DDEC) / 256, 256, 0, stream>>>(dec_f, (size_t)BTc * DDEC);
  k_silu_ip<<<(unsigned)(BTCc / 256), 256, 0, stream>>>(g_f, BTCc);
  k_ww<<<dim3(BTc, Cc / 256), 256, 0, stream>>>(dec_f, dec_w2, tdecay, ww_f);
  // 7) sequential WKV6 recurrence
  k_wkv6<<<Bc * Hc, Nc, 0, stream>>>(r_f, k_f, v_f, ww_f, faaaa, wkv_f);
  // 8) GroupNorm + gate
  k_gnorm_gate<<<(BTc * Hc) / 8, 256, 0, stream>>>(wkv_f, g_f, ln_w, ln_b, og_bf);
  // 9) output projection straight into d_out
  k_gemm_bf16<<<gg, 256, 0, stream>>>(og_bf, Wo_bf, out, BTc, Cc, Cc);
}
